// CNN_48404281426071
// MI455X (gfx1250) — compile-verified, matrix operand order checked by reading, not run on hardware
//
#include <hip/hip_runtime.h>

// ---------------------------------------------------------------------------
// CDNA5 (gfx1250) fused CNN: conv5x5(+ReLU) -> weird pool -> FC1(+ReLU) -> FC2
// Matrix work on V_WMMA_F32_16X16X32_F16 (f16 in, f32 accumulate).
// ---------------------------------------------------------------------------

typedef __attribute__((ext_vector_type(16))) _Float16 v16h;
typedef __attribute__((ext_vector_type(8)))  _Float16 v8h;
typedef __attribute__((ext_vector_type(8)))  float    v8f;

#define NSAMP 8192
#define KPAD  5888   // 5880 padded to 184 * 32
#define NPAD  112    // 100 padded to 7 * 16

union Frag16 { v16h v; v8h h[2]; };

// ---------------------------------------------------------------------------
// prep: W1 [5,5,1,30] -> w1t [32 ch][32 K] f16 (zero padded),
//       W2 [5880,100] -> w2t [112][5888] f16 transposed (B columns contiguous)
// ---------------------------------------------------------------------------
__global__ void prep_kernel(const float* __restrict__ W1,
                            const float* __restrict__ W2,
                            _Float16* __restrict__ w1t,
                            _Float16* __restrict__ w2t) {
  const int idx = blockIdx.x * blockDim.x + threadIdx.x;
  if (idx < 1024) {
    const int ch = idx >> 5, k = idx & 31;
    const float v = (ch < 30 && k < 25) ? W1[k * 30 + ch] : 0.f;  // W1[(dy*5+dx)*30 + ch]
    w1t[ch * 32 + k] = (_Float16)v;
  }
  const int j = idx - 1024;
  if (j >= 0 && j < NPAD * KPAD) {
    const int col = j / KPAD, k = j % KPAD;
    const float v = (col < 100 && k < 5880) ? W2[k * 100 + col] : 0.f;
    w2t[(size_t)col * KPAD + k] = (_Float16)v;
  }
}

// ---------------------------------------------------------------------------
// conv_pool: implicit GEMM conv (M=pixels, K=25->32, N=30->32) + ReLU + pool.
// One workgroup (8 waves) per sample.
//  * Image staged in LDS with a zero ring (35x32, stride 32) -> im2col gather
//    is 16 unconditional ds_load_b32 per lane, no bounds branches. K=25..31
//    A-lanes read harmless finite data; B columns there are zero-padded.
//  * Pool: x pairs = adjacent C-frag VGPRs (in-lane max); channel pairs =
//    adjacent lanes (shfl_xor 1).
//  * Pooled row built in LDS, then copied to global as coalesced b128.
// ---------------------------------------------------------------------------
__global__ void __launch_bounds__(256)
conv_pool_kernel(const float* __restrict__ X,
                 const _Float16* __restrict__ w1t,
                 const float* __restrict__ b1,
                 _Float16* __restrict__ pooled) {
  __shared__ float simgp[35 * 32];                 // zero-padded image
  __shared__ __align__(16) _Float16 pout[KPAD];    // pooled row staging
  const int n   = blockIdx.x;
  const int tid = threadIdx.x;

  for (int i = tid; i < 35 * 32; i += 256) simgp[i] = 0.f;
  __syncthreads();
  for (int i = tid; i < 784; i += 256) {
    const int y = i / 28, x = i % 28;
    simgp[(y + 2) * 32 + (x + 2)] = X[(size_t)n * 784 + i];
  }
  if (tid < 8) pout[5880 + tid] = (_Float16)0.f;   // K padding of this row
  __syncthreads();

  const int lane = tid & 31, wave = tid >> 5;
  const int il = lane & 15, hh = lane >> 4;

  // Tile-invariant im2col offsets for this lane's K slots
  int koff[16];
#pragma unroll
  for (int i = 0; i < 16; ++i) {
    const int k = (i < 8) ? (8 * hh + i) : (8 + 8 * hh + i);  // = 16+8*hh+(i-8)
    koff[i] = (k / 5) * 32 + (k % 5);
  }

  // B fragments (loop-invariant): columns = filter channels
  Frag16 bf0, bf1;
  {
    const _Float16* r0 = w1t + il * 32;
    const _Float16* r1 = w1t + (16 + il) * 32;
    bf0.h[0] = *(const v8h*)(r0 + 8 * hh);
    bf0.h[1] = *(const v8h*)(r0 + 16 + 8 * hh);
    bf1.h[0] = *(const v8h*)(r1 + 8 * hh);
    bf1.h[1] = *(const v8h*)(r1 + 16 + 8 * hh);
  }
  const float bias0 = b1[il];
  const float bias1 = (16 + il) < 30 ? b1[16 + il] : 0.f;

  for (int t = wave; t < 49; t += 8) {        // 784/16 = 49 pixel tiles; wave-uniform
    const int m0 = t * 16;
    const int p  = m0 + il;                   // this lane's pixel (row M = il)
    const int base = (p / 28) * 32 + (p % 28);

    Frag16 a;
#pragma unroll
    for (int i = 0; i < 16; ++i) a.v[i] = (_Float16)simgp[base + koff[i]];

    v8f c0 = {}, c1 = {};
    c0 = __builtin_amdgcn_wmma_f32_16x16x32_f16(false, a.v, false, bf0.v,
                                                (short)0, c0, false, false);
    c1 = __builtin_amdgcn_wmma_f32_16x16x32_f16(false, a.v, false, bf1.v,
                                                (short)0, c1, false, false);

    // bias + ReLU + x-pool (VGPR pairs) + channel-pool (lane pairs) -> LDS
#pragma unroll
    for (int u = 0; u < 4; ++u) {
      float q0 = fmaxf(fmaxf(c0[2 * u] + bias0, 0.f), fmaxf(c0[2 * u + 1] + bias0, 0.f));
      float q1 = fmaxf(fmaxf(c1[2 * u] + bias1, 0.f), fmaxf(c1[2 * u + 1] + bias1, 0.f));
      q0 = fmaxf(q0, __shfl_xor(q0, 1, 32));
      q1 = fmaxf(q1, __shfl_xor(q1, 1, 32));
      if ((il & 1) == 0) {
        const int pp = (m0 >> 1) + u + 4 * hh;   // pooled (y,wx) flat index
        const int cl = il >> 1;                  // pooled channel 0..7
        pout[pp * 15 + cl] = (_Float16)q0;       // channels 0..7
        if (cl < 7) pout[pp * 15 + 8 + cl] = (_Float16)q1;  // channels 8..14
      }
    }
  }
  __syncthreads();

  // Coalesced b128 copy of the finished pooled row: LDS -> global
  uint4*       dst = (uint4*)(pooled + (size_t)n * KPAD);
  const uint4* src = (const uint4*)pout;
  for (int i = tid; i < KPAD / 8; i += 256) dst[i] = src[i];   // 736 x b128
}

// ---------------------------------------------------------------------------
// fc1: [8192, 5888] @ w2t^T -> [8192, 112] f32, +bias, ReLU.
// Each wave: 32x16 output tile (2 C frags share 1 B frag), 184 K-steps.
// All 8 waves of a block share the same B tile (n0) for L1 reuse.
// ---------------------------------------------------------------------------
__global__ void __launch_bounds__(256)
fc1_kernel(const _Float16* __restrict__ pooled,
           const _Float16* __restrict__ w2t,
           const float* __restrict__ b2,
           float* __restrict__ hbuf) {
  const int tid  = threadIdx.x;
  const int lane = tid & 31, wave = tid >> 5;
  const int il = lane & 15, hh = lane >> 4;

  const int nt = blockIdx.x % 7;                  // 224 blocks = 7 n-tiles * 32
  const int mg = (blockIdx.x / 7) * 8 + wave;     // 256 m-groups of 32 rows
  const int m0 = mg * 32, n0 = nt * 16;

  const _Float16* rB  = w2t    + (size_t)(n0 + il) * KPAD;
  const _Float16* rA0 = pooled + (size_t)(m0 + il) * KPAD;
  const _Float16* rA1 = pooled + (size_t)(m0 + 16 + il) * KPAD;
  const int o1 = 8 * hh, o2 = 16 + 8 * hh;

  v8f c0 = {}, c1 = {};
#pragma unroll 2
  for (int k0 = 0; k0 < KPAD; k0 += 32) {
    Frag16 b, a0, a1;
    b.h[0]  = *(const v8h*)(rB  + k0 + o1);
    b.h[1]  = *(const v8h*)(rB  + k0 + o2);
    a0.h[0] = *(const v8h*)(rA0 + k0 + o1);
    a0.h[1] = *(const v8h*)(rA0 + k0 + o2);
    a1.h[0] = *(const v8h*)(rA1 + k0 + o1);
    a1.h[1] = *(const v8h*)(rA1 + k0 + o2);
    c0 = __builtin_amdgcn_wmma_f32_16x16x32_f16(false, a0.v, false, b.v,
                                                (short)0, c0, false, false);
    c1 = __builtin_amdgcn_wmma_f32_16x16x32_f16(false, a1.v, false, b.v,
                                                (short)0, c1, false, false);
  }

  const int col = n0 + il;
  const float bias = (col < 100) ? b2[col] : 0.f;
#pragma unroll
  for (int j = 0; j < 8; ++j) {
    const int r = m0 + j + 8 * hh;
    hbuf[(size_t)r * NPAD + col]        = fmaxf(c0[j] + bias, 0.f);
    hbuf[(size_t)(r + 16) * NPAD + col] = fmaxf(c1[j] + bias, 0.f);
  }
}

// ---------------------------------------------------------------------------
// fc2: [8192,100] @ [100,10] + b3 -> d_out [8192,10] f32. Tiny; VALU.
// ---------------------------------------------------------------------------
__global__ void fc2_kernel(const float* __restrict__ hbuf,
                           const float* __restrict__ W3,
                           const float* __restrict__ b3,
                           float* __restrict__ out) {
  const int idx = blockIdx.x * blockDim.x + threadIdx.x;
  if (idx >= NSAMP * 10) return;
  const int n = idx / 10, o = idx % 10;
  const float* hr = hbuf + (size_t)n * NPAD;
  float acc = b3[o];
#pragma unroll 4
  for (int k = 0; k < 100; ++k) acc += hr[k] * W3[k * 10 + o];
  out[idx] = acc;
}

// ---------------------------------------------------------------------------
extern "C" void kernel_launch(void* const* d_in, const int* in_sizes, int n_in,
                              void* d_out, int out_size, void* d_ws, size_t ws_size,
                              hipStream_t stream) {
  (void)in_sizes; (void)n_in; (void)out_size; (void)ws_size;
  const float* X  = (const float*)d_in[0];
  const float* W1 = (const float*)d_in[1];
  const float* b1 = (const float*)d_in[2];
  const float* W2 = (const float*)d_in[3];
  const float* b2 = (const float*)d_in[4];
  const float* W3 = (const float*)d_in[5];
  const float* b3 = (const float*)d_in[6];

  // Workspace layout (all 16-B aligned):
  //   w1t    : 32*32 f16                  =      2,048 B
  //   w2t    : 112*5888 f16               =  1,318,912 B
  //   pooled : 8192*5888 f16              = 96,468,992 B (fits in 192 MB L2)
  //   hbuf   : 8192*112 f32               =  3,670,016 B
  char* ws = (char*)d_ws;
  _Float16* w1t    = (_Float16*)ws;
  _Float16* w2t    = (_Float16*)(ws + 2048);
  _Float16* pooled = (_Float16*)(ws + 2048 + (size_t)NPAD * KPAD * 2);
  float*    hbuf   = (float*)(ws + 2048 + (size_t)NPAD * KPAD * 2
                                 + (size_t)NSAMP * KPAD * 2);

  const int prepTot = 1024 + NPAD * KPAD;
  prep_kernel<<<(prepTot + 255) / 256, 256, 0, stream>>>(W1, W2, w1t, w2t);
  conv_pool_kernel<<<NSAMP, 256, 0, stream>>>(X, w1t, b1, pooled);
  fc1_kernel<<<224, 256, 0, stream>>>(pooled, w2t, b2, hbuf);
  fc2_kernel<<<(NSAMP * 10 + 255) / 256, 256, 0, stream>>>(hbuf, W3, b3, (float*)d_out);
}
